// MPTAttention_15556371546656
// MI455X (gfx1250) — matrix operand order repaired and checked by reading, not compiled, not run on hardware
//
#include <hip/hip_runtime.h>
#include <hip/hip_bf16.h>
#include <stdint.h>

// MPT attention block for MI455X (gfx1250, wave32, WMMA).
// Matmuls: v_wmma_f32_16x16x32_bf16.
// GEMM A tiles: TDM tensor_load_to_lds (TENSORcnt), B tiles:
// global_load_async_to_lds_b128 (ASYNCcnt), double buffered.

#define S_LEN   2048
#define DMODEL  4096
#define E_QKV   12288
#define NHEADS  32
#define HDIM    128

typedef __attribute__((ext_vector_type(16))) __bf16 v16bf;
typedef __attribute__((ext_vector_type(8)))  float  v8f;
typedef unsigned int su32x4 __attribute__((ext_vector_type(4)));
typedef unsigned int su32x8 __attribute__((ext_vector_type(8)));

union Frag32 { v16bf v; uint4 u[2]; };

__device__ __forceinline__ unsigned short f2bf(float f) {
  uint32_t u = __float_as_uint(f);
  uint32_t r = u + 0x7FFFu + ((u >> 16) & 1u);   // round-to-nearest-even
  return (unsigned short)(r >> 16);
}

__device__ __forceinline__ v8f wmma_bf16(v16bf a, v16bf b, v8f c) {
  return __builtin_amdgcn_wmma_f32_16x16x32_bf16(false, a, false, b, (short)0, c, false, false);
}

__device__ __forceinline__ void async_cp16(uint32_t lds_byte_addr, const void* base, uint32_t byte_off) {
  // GVS form: mem = SADDR(u64) + VADDR(i32); LDS dest addr in VGPR. ASYNCcnt tracked.
  asm volatile("global_load_async_to_lds_b128 %0, %1, %2"
               :: "v"(lds_byte_addr), "v"(byte_off), "s"(base)
               : "memory");
}
__device__ __forceinline__ void wait_async0() {
  asm volatile("s_wait_asynccnt 0" ::: "memory");
}
__device__ __forceinline__ void wait_ds0() {
  asm volatile("s_wait_dscnt 0" ::: "memory");
}
__device__ __forceinline__ uint32_t lds_addr(const void* p) {
  return (uint32_t)(uintptr_t)p;   // low 32 bits of generic shared ptr == LDS offset
}

// ----- Tensor Data Mover: 2D tile DMA global -> LDS -------------------------
struct TdmDesc { su32x4 g0; su32x8 g1; };

// 2D bf16 tile load: tile_dim0 = 32 elems (contiguous), tile_dim1 = 128 rows,
// row pitch = pitch_elems, tensor bound tensor_rows x pitch_elems.
__device__ __forceinline__ TdmDesc make_tdm_2d(const void* gbase, uint64_t elem_off,
                                               uint32_t ldsoff, int pitch_elems,
                                               int tensor_rows) {
  TdmDesc d;
  const uint64_t ga = (uint64_t)gbase + elem_off * 2u;
  d.g0[0] = 1u;                                            // count=1 (valid user D#)
  d.g0[1] = ldsoff;                                        // lds_addr
  d.g0[2] = (uint32_t)ga;                                  // global_addr[31:0]
  d.g0[3] = (uint32_t)((ga >> 32) & 0x01FFFFFFu) | (2u << 30); // addr[56:32] | type=2
  d.g1[0] = 1u << 16;                                      // data_size=1 (2 bytes)
  d.g1[1] = ((uint32_t)pitch_elems & 0xFFFFu) << 16;       // tensor_dim0[15:0]
  d.g1[2] = (((uint32_t)pitch_elems >> 16) & 0xFFFFu)      // tensor_dim0[31:16]
          | (((uint32_t)tensor_rows & 0xFFFFu) << 16);     // tensor_dim1[15:0]
  d.g1[3] = (((uint32_t)tensor_rows >> 16) & 0xFFFFu)      // tensor_dim1[31:16]
          | (32u << 16);                                   // tile_dim0 = 32
  d.g1[4] = 128u;                                          // tile_dim1=128, tile_dim2=0
  d.g1[5] = (uint32_t)pitch_elems;                         // tensor_dim0_stride[31:0]
  d.g1[6] = 0u;                                            // stride0 hi | stride1 lo
  d.g1[7] = 0u;                                            // stride1 hi
  return d;
}

__device__ __forceinline__ void tdm_load(su32x4 g0, su32x8 g1) {
  asm volatile("tensor_load_to_lds %0, %1" :: "s"(g0), "s"(g1) : "memory");
}

// ---------------------------------------------------------------- cvt f32->bf16
__global__ void __launch_bounds__(256) cvt_bf16_kernel(const float* __restrict__ src,
                                                       unsigned short* __restrict__ dst, int n) {
  int i = blockIdx.x * 256 + threadIdx.x;
  int stride = gridDim.x * 256;
  for (; i < n; i += stride) dst[i] = f2bf(src[i]);
}

// ---------------------------------------------------------------- bf16 GEMM
// C[M,N] = A[M,K] * B[N,K]^T ; block tile 128x128, BK=32, 8 waves (16 M-rows each).
// MODE 0: plain fp32 store. MODE 1: clip +-8; col<8192 -> fp32 (stride 8192),
// col>=8192 -> bf16 V buffer (stride 4096).
template<int MODE>
__global__ void __launch_bounds__(256) gemm_bf16_kernel(
    const unsigned short* __restrict__ A, const unsigned short* __restrict__ B,
    float* __restrict__ Cf, unsigned short* __restrict__ Cv,
    int M, int N, int K) {
  __shared__ __align__(16) unsigned short lA[2][128 * 32];
  __shared__ __align__(16) unsigned short lB[2][128 * 32];
  const int t = threadIdx.x;
  const int w = t >> 5, lane = t & 31;
  const int m0 = blockIdx.y * 128, n0 = blockIdx.x * 128;
  const int nl  = lane & 15;              // A: M row within 16 / B: N col within 16
  const int hi  = lane & 16;

  v8f acc[8];
#pragma unroll
  for (int j = 0; j < 8; ++j) acc[j] = (v8f){0.f,0.f,0.f,0.f,0.f,0.f,0.f,0.f};

  // A tile via Tensor Data Mover (single DMA, wave 0 only; EXEC ignored by TDM)
  auto stage_A = [&](int bufi, int k0) {
    if (w == 0) {
      TdmDesc d = make_tdm_2d(A, (uint64_t)m0 * K + k0, lds_addr(&lA[bufi][0]), K, M);
      tdm_load(d.g0, d.g1);
    }
  };
  // B tile via per-lane async copies (all 256 threads)
  auto stage_B = [&](int bufi, int k0) {
#pragma unroll
    for (int i = 0; i < 2; ++i) {
      int c = t + 256 * i;                // 0..511 -> 128 rows x 4 segs of 16B
      int row = c >> 2, seg = c & 3;
      async_cp16(lds_addr(&lB[bufi][row * 32 + seg * 8]), B,
                 (uint32_t)(((size_t)(n0 + row) * K + k0 + seg * 8) * 2));
    }
  };
  auto wait_stage = [&]() {
    wait_async0();                               // each wave: own async copies done
    if (w == 0) __builtin_amdgcn_s_wait_tensorcnt(0); // wave 0: TDM done
    __syncthreads();
  };

  stage_A(0, 0);
  stage_B(0, 0);
  wait_stage();

  int buf = 0;
  const int ksteps = K >> 5;
  for (int kt = 0; kt < ksteps; ++kt) {
    if (kt + 1 < ksteps) {
      stage_A(buf ^ 1, (kt + 1) * 32);
      stage_B(buf ^ 1, (kt + 1) * 32);
    }

    // A fragment: lane<16 holds K {0..15}, lane>=16 holds {16..31}
    Frag32 af;
    {
      const unsigned short* p = &lA[buf][(w * 16 + nl) * 32 + (hi ? 16 : 0)];
      af.u[0] = *(const uint4*)p;
      af.u[1] = *(const uint4*)(p + 8);
    }
#pragma unroll
    for (int j = 0; j < 8; ++j) {
      Frag32 bf_;
      const unsigned short* p = &lB[buf][(j * 16 + nl) * 32 + (hi ? 16 : 0)];
      bf_.u[0] = *(const uint4*)p;
      bf_.u[1] = *(const uint4*)(p + 8);
      acc[j] = wmma_bf16(af.v, bf_.v, acc[j]);
    }

    wait_stage();
    buf ^= 1;
  }

#pragma unroll
  for (int j = 0; j < 8; ++j) {
#pragma unroll
    for (int e = 0; e < 8; ++e) {
      int row = m0 + w * 16 + e + (hi ? 8 : 0);
      int col = n0 + j * 16 + nl;
      float v = acc[j][e];
      if (MODE == 1) {
        v = fminf(fmaxf(v, -8.f), 8.f);
        if (col < 8192) Cf[(size_t)row * 8192 + col] = v;
        else            Cv[(size_t)row * 4096 + (col - 8192)] = f2bf(v);
      } else {
        Cf[(size_t)row * N + col] = v;
      }
    }
  }
}

// ---------------------------------------------------------------- LayerNorm q/k
__global__ void __launch_bounds__(256) ln_kernel(
    const float* __restrict__ qk,
    const float* __restrict__ qw, const float* __restrict__ qbias,
    const float* __restrict__ kw, const float* __restrict__ kbias,
    unsigned short* __restrict__ Qb, unsigned short* __restrict__ Kb) {
  const int b = blockIdx.x;
  const int row = b >> 1, which = b & 1;
  const float* src = qk + (size_t)row * 8192 + which * 4096;
  const float* w   = which ? kw : qw;
  const float* bia = which ? kbias : qbias;
  unsigned short* dst = (which ? Kb : Qb) + (size_t)row * 4096;

  float s = 0.f, s2 = 0.f;
  for (int i = threadIdx.x; i < 4096; i += 256) {
    float x = src[i]; s += x; s2 += x * x;
  }
  for (int m = 16; m >= 1; m >>= 1) { s += __shfl_xor(s, m); s2 += __shfl_xor(s2, m); }
  __shared__ float rs[8], rs2[8];
  if ((threadIdx.x & 31) == 0) { rs[threadIdx.x >> 5] = s; rs2[threadIdx.x >> 5] = s2; }
  __syncthreads();
  s = 0.f; s2 = 0.f;
#pragma unroll
  for (int i = 0; i < 8; ++i) { s += rs[i]; s2 += rs2[i]; }
  const float mean = s * (1.f / 4096.f);
  const float var  = s2 * (1.f / 4096.f) - mean * mean;
  const float rstd = rsqrtf(var + 1e-5f);
  for (int i = threadIdx.x; i < 4096; i += 256)
    dst[i] = f2bf((src[i] - mean) * rstd * w[i] + bia[i]);
}

// ---------------------------------------------------------------- V transpose
// Vb[2048][4096] -> Vt[4096][2048]  (so PV B-fragments are contiguous along kv)
__global__ void __launch_bounds__(256) transpose_v_kernel(
    const unsigned short* __restrict__ Vb, unsigned short* __restrict__ Vt) {
  __shared__ __align__(16) unsigned short tile[32][33];
  const int dBase = blockIdx.x * 32;
  const int sBase = blockIdx.y * 32;
  const int tx = threadIdx.x & 31, ty = threadIdx.x >> 5;
#pragma unroll
  for (int i = 0; i < 32; i += 8)
    tile[ty + i][tx] = Vb[(size_t)(sBase + ty + i) * 4096 + dBase + tx];
  __syncthreads();
#pragma unroll
  for (int i = 0; i < 32; i += 8)
    Vt[(size_t)(dBase + ty + i) * 2048 + sBase + tx] = tile[tx][ty + i];
}

// ---------------------------------------------------------------- flash attention
// grid(32 heads, 16); 8 waves/block, each wave owns 16 q rows; kv blocks of 32.
__global__ void __launch_bounds__(256) attn_kernel(
    const unsigned short* __restrict__ Qb, const unsigned short* __restrict__ Kb,
    const unsigned short* __restrict__ Vt, unsigned short* __restrict__ Ob) {
  __shared__ __align__(16) unsigned short pbuf[8][16 * 32];
  const int h = blockIdx.x;
  const int w = threadIdx.x >> 5, lane = threadIdx.x & 31;
  const int q0   = (blockIdx.y * 8 + w) * 16;
  const int nl   = lane & 15;
  const int hi8  = (lane & 16) ? 8 : 0;    // C row offset
  const int hi16 = (lane & 16) ? 16 : 0;   // B-frag K base
  const float scale = 0.08838834764831845f;                 // 1/sqrt(128)
  const float slope = exp2f(-0.25f * (float)(h + 1));       // ALiBi slope

  Frag32 qf[4];
#pragma unroll
  for (int c = 0; c < 4; ++c) {
    const unsigned short* p = Qb + (size_t)(q0 + nl) * 4096 + h * 128 + c * 32 + hi16;
    qf[c].u[0] = *(const uint4*)p;
    qf[c].u[1] = *(const uint4*)(p + 8);
  }

  v8f o[8];
#pragma unroll
  for (int c = 0; c < 8; ++c) o[c] = (v8f){0.f,0.f,0.f,0.f,0.f,0.f,0.f,0.f};
  float mrow[8], lrow[8];
#pragma unroll
  for (int j = 0; j < 8; ++j) { mrow[j] = -1e30f; lrow[j] = 0.f; }

  const int kvblocks = (q0 + 47) >> 5;     // cover kv <= q0+15
  for (int b = 0; b < kvblocks; ++b) {
    const int kv0 = b * 32;
    v8f s0 = (v8f){0.f,0.f,0.f,0.f,0.f,0.f,0.f,0.f};
    v8f s1 = s0;
#pragma unroll
    for (int c = 0; c < 4; ++c) {
      Frag32 k0f, k1f;
      const unsigned short* p0 = Kb + (size_t)(kv0 + nl) * 4096 + h * 128 + c * 32 + hi16;
      k0f.u[0] = *(const uint4*)p0; k0f.u[1] = *(const uint4*)(p0 + 8);
      const unsigned short* p1 = Kb + (size_t)(kv0 + 16 + nl) * 4096 + h * 128 + c * 32 + hi16;
      k1f.u[0] = *(const uint4*)p1; k1f.u[1] = *(const uint4*)(p1 + 8);
      s0 = wmma_bf16(qf[c].v, k0f.v, s0);
      s1 = wmma_bf16(qf[c].v, k1f.v, s1);
    }

    float p0v[8], p1v[8], fac[8];
#pragma unroll
    for (int j = 0; j < 8; ++j) {
      const float qp  = (float)(q0 + j + hi8);
      const float kp0 = (float)(kv0 + nl);
      const float kp1 = kp0 + 16.f;
      float v0 = s0[j] * scale + slope * (kp0 - qp);
      float v1 = s1[j] * scale + slope * (kp1 - qp);
      if (kp0 > qp) v0 = -1e30f;
      if (kp1 > qp) v1 = -1e30f;
      float mx = fmaxf(v0, v1);
      mx = fmaxf(mx, __shfl_xor(mx, 1));
      mx = fmaxf(mx, __shfl_xor(mx, 2));
      mx = fmaxf(mx, __shfl_xor(mx, 4));
      mx = fmaxf(mx, __shfl_xor(mx, 8));
      const float mnew = fmaxf(mrow[j], mx);
      const float f = __expf(mrow[j] - mnew);
      const float e0 = __expf(v0 - mnew);
      const float e1 = __expf(v1 - mnew);
      float sm = e0 + e1;
      sm += __shfl_xor(sm, 1);
      sm += __shfl_xor(sm, 2);
      sm += __shfl_xor(sm, 4);
      sm += __shfl_xor(sm, 8);
      lrow[j] = lrow[j] * f + sm;
      mrow[j] = mnew;
      fac[j] = f; p0v[j] = e0; p1v[j] = e1;
    }
#pragma unroll
    for (int c = 0; c < 8; ++c)
#pragma unroll
      for (int j = 0; j < 8; ++j) o[c][j] *= fac[j];

    // C-layout P (16x32) -> LDS -> reload in A-fragment layout
#pragma unroll
    for (int j = 0; j < 8; ++j) {
      const int row = j + hi8;
      pbuf[w][row * 32 + nl]      = f2bf(p0v[j]);
      pbuf[w][row * 32 + 16 + nl] = f2bf(p1v[j]);
    }
    wait_ds0();
    Frag32 pf;
    {
      const unsigned short* p = &pbuf[w][nl * 32 + hi16];
      pf.u[0] = *(const uint4*)p;
      pf.u[1] = *(const uint4*)(p + 8);
    }
#pragma unroll
    for (int c = 0; c < 8; ++c) {
      Frag32 vf;
      const unsigned short* p = Vt + (size_t)(h * 128 + c * 16 + nl) * 2048 + kv0 + hi16;
      vf.u[0] = *(const uint4*)p; vf.u[1] = *(const uint4*)(p + 8);
      o[c] = wmma_bf16(pf.v, vf.v, o[c]);
    }
  }

#pragma unroll
  for (int c = 0; c < 8; ++c)
#pragma unroll
    for (int j = 0; j < 8; ++j) {
      const float inv = __builtin_amdgcn_rcpf(lrow[j]);
      Ob[(size_t)(q0 + j + hi8) * 4096 + h * 128 + c * 16 + nl] = f2bf(o[c][j] * inv);
    }
}

// ---------------------------------------------------------------- launch
extern "C" void kernel_launch(void* const* d_in, const int* in_sizes, int n_in,
                              void* d_out, int out_size, void* d_ws, size_t ws_size,
                              hipStream_t stream) {
  (void)in_sizes; (void)n_in; (void)out_size; (void)ws_size;
  const float* hidden = (const float*)d_in[0];
  const float* Wqkv   = (const float*)d_in[1];
  const float* q_w    = (const float*)d_in[2];
  const float* q_b    = (const float*)d_in[3];
  const float* k_w    = (const float*)d_in[4];
  const float* k_b    = (const float*)d_in[5];
  const float* Wout   = (const float*)d_in[6];
  float* out = (float*)d_out;

  char* ws = (char*)d_ws;
  size_t off = 0;
  auto carve = [&](size_t bytes) -> void* {
    void* p = ws + off;
    off = (off + bytes + 255) & ~(size_t)255;
    return p;
  };
  unsigned short* Hb     = (unsigned short*)carve((size_t)S_LEN * DMODEL * 2);
  unsigned short* Wqkvb  = (unsigned short*)carve((size_t)E_QKV * DMODEL * 2);
  unsigned short* Woutb  = (unsigned short*)carve((size_t)DMODEL * DMODEL * 2);
  float*          QKf    = (float*)carve((size_t)S_LEN * 8192 * 4);
  unsigned short* Vb     = (unsigned short*)carve((size_t)S_LEN * DMODEL * 2);
  unsigned short* Qb     = (unsigned short*)carve((size_t)S_LEN * DMODEL * 2);
  unsigned short* Kb     = (unsigned short*)carve((size_t)S_LEN * DMODEL * 2);
  unsigned short* Vt     = (unsigned short*)carve((size_t)DMODEL * S_LEN * 2);
  unsigned short* Ob     = (unsigned short*)carve((size_t)S_LEN * DMODEL * 2);

  cvt_bf16_kernel<<<4096, 256, 0, stream>>>(hidden, Hb, S_LEN * DMODEL);
  cvt_bf16_kernel<<<4096, 256, 0, stream>>>(Wqkv, Wqkvb, E_QKV * DMODEL);
  cvt_bf16_kernel<<<4096, 256, 0, stream>>>(Wout, Woutb, DMODEL * DMODEL);

  gemm_bf16_kernel<1><<<dim3(E_QKV / 128, S_LEN / 128), 256, 0, stream>>>(
      Hb, Wqkvb, QKf, Vb, S_LEN, E_QKV, DMODEL);

  ln_kernel<<<S_LEN * 2, 256, 0, stream>>>(QKf, q_w, q_b, k_w, k_b, Qb, Kb);

  transpose_v_kernel<<<dim3(DMODEL / 32, S_LEN / 32), 256, 0, stream>>>(Vb, Vt);

  attn_kernel<<<dim3(NHEADS, S_LEN / 16 / 8), 256, 0, stream>>>(Qb, Kb, Vt, Ob);

  gemm_bf16_kernel<0><<<dim3(DMODEL / 128, S_LEN / 128), 256, 0, stream>>>(
      Ob, Woutb, out, nullptr, S_LEN, DMODEL, DMODEL);
}
